// MultipleHeadSelfAttention_39745627357249
// MI455X (gfx1250) — matrix operand order repaired and checked by reading, not compiled, not run on hardware
//
#include <hip/hip_runtime.h>

typedef _Float16 half_t;
typedef __attribute__((ext_vector_type(16))) _Float16 v16h;
typedef __attribute__((ext_vector_type(8)))  _Float16 v8h;
typedef __attribute__((ext_vector_type(4)))  _Float16 v4h;
typedef __attribute__((ext_vector_type(8)))  float    v8f;
typedef __attribute__((ext_vector_type(4)))  float    v4f;
typedef __attribute__((ext_vector_type(4)))  unsigned int v4u;
typedef __attribute__((ext_vector_type(4)))  int      v4i;
typedef __attribute__((ext_vector_type(8)))  int      v8i;

#define WMMA_F16(a, b, c) \
    __builtin_amdgcn_wmma_f32_16x16x32_f16(false, (a), false, (b), (short)0, (c), false, false)

// Problem constants
constexpr int Bq   = 128;
constexpr int Tq   = 256;
constexpr int Cq   = 384;
constexpr int Hq   = 6;
constexpr int Dq   = 64;
constexpr int NQKV = 3 * Cq;        // 1152
constexpr int BT   = Bq * Tq;       // 32768

// ---------------------------------------------------------------------------
// TDM: 1-D contiguous f16 tile, global -> LDS.  D# per CDNA5 ISA.
// 6-arg builtin (clang-23 / therock form): (v4u, v8i, v4i, v4i, v8i, i32 cpol)
// ---------------------------------------------------------------------------
#if __has_builtin(__builtin_amdgcn_tensor_load_to_lds)
#define HAVE_TDM 1
__device__ __forceinline__ unsigned rfl(unsigned v) {
    return (unsigned)__builtin_amdgcn_readfirstlane((int)v);
}
__device__ __forceinline__ void tdm_load_1d_f16(const half_t* g, void* lds, int nelems)
{
    unsigned long long ga = (unsigned long long)(size_t)g;
    unsigned lds_off = rfl((unsigned)(size_t)lds);   // LDS aperture: addr[31:0]
    v4u g0;
    g0.x = 1u;                                        // count=1, no gather
    g0.y = lds_off;                                   // lds_addr
    g0.z = rfl((unsigned)ga);                         // global_addr[31:0]
    g0.w = rfl((unsigned)((ga >> 32) & 0x01FFFFFFu)) | (2u << 30);  // [56:32] | type=2
    unsigned ne = (unsigned)nelems;
    v8i g1;
    g1[0] = (int)rfl(1u << 16);                       // mask=0, data_size=2B
    g1[1] = (int)rfl(ne << 16);                       // tensor_dim0[15:0]
    g1[2] = (int)rfl((ne >> 16) | (1u << 16));        // tensor_dim0[31:16] | tensor_dim1=1
    g1[3] = (int)rfl((ne & 0xFFFFu) << 16);           // tile_dim0
    g1[4] = (int)rfl(1u);                             // tile_dim1=1, tile_dim2=0
    g1[5] = (int)rfl(ne);                             // tensor_dim0_stride
    g1[6] = 0;
    g1[7] = 0;
    v4i zz4 = {0, 0, 0, 0};
    v8i zz8 = {0, 0, 0, 0, 0, 0, 0, 0};
    __builtin_amdgcn_tensor_load_to_lds(g0, g1, zz4, zz4, zz8, 0);
}
#endif

// ---------------------------------------------------------------------------
// Kernel 1: f16-convert x, transpose + f16-convert the weight matrices.
//   Xh[i]       = (f16) x[i]        (BT x 384, 4 elems/thread)
//   WqkvT[n][k] = W_qkv[k][n]       (1152 x 384)
//   WprojT[n][k] = W_proj[k][n]     (384 x 384)
// ---------------------------------------------------------------------------
__global__ __launch_bounds__(256) void convert_kernel(
    const float* __restrict__ x, const float* __restrict__ Wqkv,
    const float* __restrict__ Wproj,
    half_t* __restrict__ Xh, half_t* __restrict__ WqkvT, half_t* __restrict__ WprojT)
{
    int i = blockIdx.x * blockDim.x + threadIdx.x;
    // x conversion: 12582912 elems, 4 per thread
    {
        size_t i4 = (size_t)i * 4;
        v4f f = *(const v4f*)(x + i4);
        v4h h = { (half_t)f.x, (half_t)f.y, (half_t)f.z, (half_t)f.w };
        *(v4h*)(Xh + i4) = h;
    }
    if (i < NQKV * Cq) {
        int n = i / Cq, k = i % Cq;
        WqkvT[i] = (half_t)Wqkv[(size_t)k * NQKV + n];
    }
    if (i < Cq * Cq) {
        int n = i / Cq, k = i % Cq;
        WprojT[i] = (half_t)Wproj[(size_t)k * Cq + n];
    }
}

// ---------------------------------------------------------------------------
// Kernel 2: qkv = x @ W_qkv with fused RoPE on q,k; scatter into
//   Q[b][h][t][d], K[b][h][t][d], Vt[b][h][d][t]   (all f16)
// One 16x16 output tile per wave, K=384 via 12 x wmma_f32_16x16x32_f16.
// ---------------------------------------------------------------------------
__global__ __launch_bounds__(256) void qkv_rope_kernel(
    const half_t* __restrict__ Xh, const half_t* __restrict__ WqkvT,
    half_t* __restrict__ Qb, half_t* __restrict__ Kb, half_t* __restrict__ Vt)
{
    const int lane = threadIdx.x & 31;
    const int wave = threadIdx.x >> 5;
    const int l16  = lane & 15;
    const int hh   = lane >> 4;

    const int tile = blockIdx.x * 8 + wave;     // 147456 tiles total
    const int mt = tile / (NQKV / 16);
    const int nt = tile % (NQKV / 16);
    const int m0 = mt * 16;
    const int n0 = nt * 16;

    const half_t* arow = Xh    + (size_t)(m0 + l16) * Cq;
    const half_t* brow = WqkvT + (size_t)(n0 + l16) * Cq;

    v8f acc = {};
    for (int k0 = 0; k0 < Cq; k0 += 32) {
        const int a1 = k0 + hh * 8;
        __builtin_prefetch(arow + a1 + 32, 0, 0);
        const v16h a = *(const v16h*)(arow + k0 + hh * 16);
        const v16h b = *(const v16h*)(brow + k0 + hh * 16);
        acc = WMMA_F16(a, b, acc);
    }

    // Decode destination region.
    const int batch  = m0 / Tq;
    const int t0     = m0 % Tq;
    const int region = n0 / Cq;              // 0=q 1=k 2=v
    const int cin    = n0 % Cq;
    const int head   = cin / Dq;             // constant over the tile
    const int d      = (cin % Dq) + l16;     // this lane's head-dim column

    if (region < 2) {
        // RoPE: angle = t * base^(-2*(d/2)/64); pair partner lives in lane^1.
        const float invf = __expf(-(float)((d >> 1) << 1) * (9.210340371976184f / 64.0f));
        const float sgn  = (d & 1) ? 1.0f : -1.0f;
        half_t* dst = ((region == 0) ? Qb : Kb) +
                      (((size_t)batch * Hq + head) * Tq) * Dq;
#pragma unroll
        for (int vi = 0; vi < 8; ++vi) {
            const int r = vi + 8 * hh;
            const int t = t0 + r;
            float val  = acc[vi];
            float part = __shfl_xor(val, 1, 32);
            float sn, cs;
            __sincosf((float)t * invf, &sn, &cs);
            float res = val * cs + sgn * part * sn;
            dst[(size_t)t * Dq + d] = (half_t)res;
        }
    } else {
        half_t* dst = Vt + ((((size_t)batch * Hq + head) * Dq + d) * Tq);
#pragma unroll
        for (int vi = 0; vi < 8; ++vi) {
            const int t = t0 + vi + 8 * hh;
            dst[t] = (half_t)acc[vi];
        }
    }
}

// ---------------------------------------------------------------------------
// Kernel 3: attention for one (b,h) per block. 512 threads = 16 waves,
// wave w owns query rows [16w, 16w+16). Causal: only key tiles 0..(w|1).
// LDS: K [256][64] + Vt [64][256] (TDM-loaded) + per-wave P strip (f16).
// ---------------------------------------------------------------------------
__global__ __launch_bounds__(512) void attn_kernel(
    const half_t* __restrict__ Qb, const half_t* __restrict__ Kb,
    const half_t* __restrict__ Vt, half_t* __restrict__ Attn)
{
    extern __shared__ char smem[];
    half_t* Ksh = (half_t*)smem;            // 256*64
    half_t* Vsh = Ksh + Tq * Dq;            // 64*256
    half_t* Psh = Vsh + Dq * Tq;            // 16 waves * 16 * 256

    const int bh = blockIdx.x;
    const half_t* Qg = Qb + (size_t)bh * Tq * Dq;
    const half_t* Kg = Kb + (size_t)bh * Tq * Dq;
    const half_t* Vg = Vt + (size_t)bh * Dq * Tq;

#ifdef HAVE_TDM
    // Tensor Data Mover: bulk K and V^T tiles into LDS, issued by wave 0.
    if (threadIdx.x < 32) {
        tdm_load_1d_f16(Kg, Ksh, Tq * Dq);
        tdm_load_1d_f16(Vg, Vsh, Dq * Tq);
        __builtin_amdgcn_s_wait_tensorcnt(0);
    }
    __syncthreads();
#else
    for (int i = threadIdx.x; i < (Tq * Dq) / 8; i += 512) {
        ((v8h*)Ksh)[i] = ((const v8h*)Kg)[i];
        ((v8h*)Vsh)[i] = ((const v8h*)Vg)[i];
    }
    __syncthreads();
#endif

    const int lane = threadIdx.x & 31;
    const int w    = threadIdx.x >> 5;
    const int l16  = lane & 15;
    const int hh   = lane >> 4;
    const int q0   = w * 16;

    // Q A-fragments for this wave's 16 rows (two 32-wide d-chunks).
    v16h qf[2];
    {
        const half_t* qrow = Qg + (size_t)(q0 + l16) * Dq;
#pragma unroll
        for (int kk = 0; kk < 2; ++kk) {
            qf[kk] = *(const v16h*)(qrow + kk * 32 + hh * 16);
        }
    }

    const int ntiles  = (w | 1) + 1;   // key tiles needed (even count)
    const int kchunks = ntiles >> 1;   // 32-wide K chunks for PV

    // ---- scores S = Q K^T ----
    v8f s[16];
#pragma unroll
    for (int nt = 0; nt < 16; ++nt) {
        if (nt < ntiles) {
            v8f a = {};
#pragma unroll
            for (int kk = 0; kk < 2; ++kk) {
                const v16h b = *(const v16h*)(Ksh + (size_t)(nt * 16 + l16) * Dq +
                                              kk * 32 + hh * 16);
                a = WMMA_F16(qf[kk], b, a);
            }
            s[nt] = a;
        }
    }

    // ---- causal mask + two-pass softmax (rows = vi + 8*hh) ----
    float rmax[8], rsum[8];
#pragma unroll
    for (int vi = 0; vi < 8; ++vi) { rmax[vi] = -1e30f; rsum[vi] = 0.0f; }

#pragma unroll
    for (int nt = 0; nt < 16; ++nt) {
        if (nt < ntiles) {
#pragma unroll
            for (int vi = 0; vi < 8; ++vi) {
                const int qi = q0 + vi + 8 * hh;
                const int kj = nt * 16 + l16;
                float v = s[nt][vi] * 0.125f;          // 1/sqrt(64)
                v = (kj <= qi) ? v : -1e30f;
                s[nt][vi] = v;
                rmax[vi] = fmaxf(rmax[vi], v);
            }
        }
    }
#pragma unroll
    for (int vi = 0; vi < 8; ++vi)
#pragma unroll
        for (int off = 1; off < 16; off <<= 1)
            rmax[vi] = fmaxf(rmax[vi], __shfl_xor(rmax[vi], off, 32));

#pragma unroll
    for (int nt = 0; nt < 16; ++nt) {
        if (nt < ntiles) {
#pragma unroll
            for (int vi = 0; vi < 8; ++vi) {
                float e = __expf(s[nt][vi] - rmax[vi]);
                s[nt][vi] = e;
                rsum[vi] += e;
            }
        }
    }
#pragma unroll
    for (int vi = 0; vi < 8; ++vi)
#pragma unroll
        for (int off = 1; off < 16; off <<= 1)
            rsum[vi] += __shfl_xor(rsum[vi], off, 32);

    // ---- stage P (f16) in this wave's LDS strip: C-layout -> A-layout ----
    half_t* pstrip = Psh + (size_t)w * 16 * Tq;
#pragma unroll
    for (int nt = 0; nt < 16; ++nt) {
        if (nt < ntiles) {
#pragma unroll
            for (int vi = 0; vi < 8; ++vi) {
                const int r = vi + 8 * hh;
                pstrip[r * Tq + nt * 16 + l16] = (half_t)s[nt][vi];
            }
        }
    }
    asm volatile("s_wait_dscnt 0x0" ::: "memory");

    // ---- O = P V  (K = up to 256, chunks of 32) ----
    v8f o[4] = {{}, {}, {}, {}};
    const half_t* prow = pstrip + (size_t)l16 * Tq;    // A row m = l16
#pragma unroll
    for (int kk = 0; kk < 8; ++kk) {
        if (kk < kchunks) {
            const v16h pa = *(const v16h*)(prow + kk * 32 + hh * 16);
#pragma unroll
            for (int dt = 0; dt < 4; ++dt) {
                const v16h b = *(const v16h*)(Vsh + (size_t)(dt * 16 + l16) * Tq +
                                              kk * 32 + hh * 16);
                o[dt] = WMMA_F16(pa, b, o[dt]);
            }
        }
    }

    // ---- normalize by rowsum, store into Attn[b][t][h*64+d] (f16) ----
    const int batch = bh / Hq;
    const int head  = bh % Hq;
    half_t* obase = Attn + (size_t)batch * Tq * Cq + head * Dq;
#pragma unroll
    for (int dt = 0; dt < 4; ++dt) {
#pragma unroll
        for (int vi = 0; vi < 8; ++vi) {
            const int r = vi + 8 * hh;
            const int t = q0 + r;
            float val = o[dt][vi] / rsum[vi];
            obase[(size_t)t * Cq + dt * 16 + l16] = (half_t)val;
        }
    }
}

// ---------------------------------------------------------------------------
// Kernel 4: out = Attn @ W_proj  (fp32 output)
// ---------------------------------------------------------------------------
__global__ __launch_bounds__(256) void proj_kernel(
    const half_t* __restrict__ Attn, const half_t* __restrict__ WprojT,
    float* __restrict__ out)
{
    const int lane = threadIdx.x & 31;
    const int wave = threadIdx.x >> 5;
    const int l16  = lane & 15;
    const int hh   = lane >> 4;

    const int tile = blockIdx.x * 8 + wave;     // 49152 tiles
    const int mt = tile / (Cq / 16);
    const int nt = tile % (Cq / 16);
    const int m0 = mt * 16;
    const int n0 = nt * 16;

    const half_t* arow = Attn   + (size_t)(m0 + l16) * Cq;
    const half_t* brow = WprojT + (size_t)(n0 + l16) * Cq;

    v8f acc = {};
    for (int k0 = 0; k0 < Cq; k0 += 32) {
        const int a1 = k0 + hh * 8;
        __builtin_prefetch(arow + a1 + 32, 0, 0);
        const v16h a = *(const v16h*)(arow + k0 + hh * 16);
        const v16h b = *(const v16h*)(brow + k0 + hh * 16);
        acc = WMMA_F16(a, b, acc);
    }
#pragma unroll
    for (int vi = 0; vi < 8; ++vi) {
        const int r = vi + 8 * hh;
        out[(size_t)(m0 + r) * Cq + n0 + l16] = acc[vi];
    }
}

// ---------------------------------------------------------------------------
extern "C" void kernel_launch(void* const* d_in, const int* in_sizes, int n_in,
                              void* d_out, int out_size, void* d_ws, size_t ws_size,
                              hipStream_t stream) {
    const float* x     = (const float*)d_in[0];
    const float* Wqkv  = (const float*)d_in[1];
    const float* Wproj = (const float*)d_in[2];
    float* out = (float*)d_out;

    // Workspace layout (f16):
    //   WqkvT 1152*384, WprojT 384*384, Xh BT*C, Q/K/Vt each B*H*T*D, Attn BT*C
    char* ws = (char*)d_ws;
    half_t* WqkvT  = (half_t*)ws;                               //   884736 B
    half_t* WprojT = (half_t*)(ws + 884736);                    //   294912 B
    half_t* Xh     = (half_t*)(ws + 1179648);                   // 25165824 B
    half_t* Qb     = (half_t*)(ws + 26345472);
    half_t* Kb     = Qb + (size_t)Bq * Hq * Tq * Dq;            // 12582912 elems each
    half_t* Vt     = Kb + (size_t)Bq * Hq * Tq * Dq;
    half_t* At     = Vt + (size_t)Bq * Hq * Tq * Dq;

    convert_kernel<<<(BT * Cq / 4) / 256, 256, 0, stream>>>(
        x, Wqkv, Wproj, Xh, WqkvT, WprojT);

    qkv_rope_kernel<<<(BT / 16) * (NQKV / 16) / 8, 256, 0, stream>>>(
        Xh, WqkvT, Qb, Kb, Vt);

    const size_t lds_bytes =
        (size_t)(Tq * Dq + Dq * Tq + 16 * 16 * Tq) * sizeof(half_t);  // 196608
    (void)hipFuncSetAttribute(reinterpret_cast<const void*>(attn_kernel),
                              hipFuncAttributeMaxDynamicSharedMemorySize,
                              (int)lds_bytes);
    attn_kernel<<<Bq * Hq, 512, lds_bytes, stream>>>(Qb, Kb, Vt, At);

    proj_kernel<<<(BT / 16) * (Cq / 16) / 8, 256, 0, stream>>>(At, WprojT, out);
}
